// RGBDPoserHeatmap_71227737637590
// MI455X (gfx1250) — matrix operand order repaired and checked by reading, not compiled, not run on hardware
//
#include <hip/hip_runtime.h>
#include <hip/hip_bf16.h>
#include <stdint.h>

// ---------------- problem constants ----------------
static constexpr int HH  = 64;     // heatmap H
static constexpr int WW  = 80;     // heatmap W
static constexpr int HW  = HH * WW;      // 5120 pixels / image
static constexpr int NB  = 64;     // batch
static constexpr int CC  = 128;    // channels per modality
static constexpr int C2  = 256;    // combined channels
static constexpr int JJ  = 22;     // joints
static constexpr int JP  = 32;     // joints padded to 2 WMMA N-tiles

// ---------------- workspace layout (bytes) ----------------
static constexpr size_t WS_LOSS   = 0;                      // 1 float (64B slot)
static constexpr size_t WS_MEANC  = 64;                     // 64*256 floats
static constexpr size_t WS_BCOMBO = WS_MEANC + 65536;       // 32 floats
static constexpr size_t WS_XC     = WS_BCOMBO + 128;        // 64*32 ints
static constexpr size_t WS_YC     = WS_XC + 8192;
static constexpr size_t WS_MASK   = WS_YC + 8192;
static constexpr size_t WS_BLOB   = WS_MASK + 8192;         // 8192 bf16 = 16 KB (16B aligned)

typedef __bf16    v16bf __attribute__((ext_vector_type(16)));
typedef float     v8f   __attribute__((ext_vector_type(8)));
typedef uint32_t  v4u   __attribute__((ext_vector_type(4)));
typedef int       v8i   __attribute__((ext_vector_type(8)));
typedef int       v4i   __attribute__((ext_vector_type(4)));

#if defined(__HIP_DEVICE_COMPILE__) && \
    __has_builtin(__builtin_amdgcn_tensor_load_to_lds) && \
    __has_builtin(__builtin_amdgcn_s_wait_tensorcnt)
#define USE_TDM 1
#else
#define USE_TDM 0
#endif

// =====================================================================
// Kernel 1: prep — zero accumulators, fuse weights (w_combo = w_hm@w_proj)
// pre-swizzled into WMMA B-operand lane order, splat centers / targets.
// =====================================================================
__global__ void __launch_bounds__(256)
posek_prep(const float* __restrict__ gt,
           const float* __restrict__ w_proj, const float* __restrict__ b_proj,
           const float* __restrict__ w_hm,   const float* __restrict__ b_hm,
           unsigned char* __restrict__ ws)
{
    const int tid = threadIdx.x;
    float* lossAcc = (float*)(ws + WS_LOSS);
    float* meanc   = (float*)(ws + WS_MEANC);
    float* bcombo  = (float*)(ws + WS_BCOMBO);
    int*   xc      = (int*)  (ws + WS_XC);
    int*   yc      = (int*)  (ws + WS_YC);
    float* msk     = (float*)(ws + WS_MASK);
    __bf16* blob   = (__bf16*)(ws + WS_BLOB);

    if (tid == 0) lossAcc[0] = 0.0f;
    for (int i = tid; i < NB * C2; i += 256) meanc[i] = 0.0f;

    // b_combo[j] = w_hm[j,:] @ b_proj + b_hm[j]
    if (tid < JP) {
        float v = 0.0f;
        if (tid < JJ) {
            v = b_hm[tid];
            for (int o = 0; o < CC; ++o) v += w_hm[tid * CC + o] * b_proj[o];
        }
        bcombo[tid] = v;
    }

    // per-(b,j) splat center (int trunc), center validity, presence target
    for (int idx = tid; idx < NB * JP; idx += 256) {
        const int b = idx >> 5, j = idx & 31;
        int cx = -1000, cy = -1000;       // sentinel => gaussian always 0
        float t = 0.0f;
        if (j < JJ) {
            const float gx = gt[(b * JJ + j) * 2 + 0];
            const float gy = gt[(b * JJ + j) * 2 + 1];
            const int x = (int)truncf(gx * 0.25f);   // * (80/320)
            const int y = (int)truncf(gy * 0.25f);   // * (64/256)
            if (x >= 0 && y >= 0 && x < WW && y < HH) { cx = x; cy = y; }
            t = (gx >= 0.0f && gx < 320.0f && gy >= 0.0f && gy < 256.0f) ? 1.0f : 0.0f;
        }
        xc[idx] = cx; yc[idx] = cy; msk[idx] = t;
    }

    // w_combo[j][c] = sum_o w_hm[j,o] * w_proj[o,c], laid out in a contiguous-k
    // 16-bit WMMA B (32x16) lane order: tile(s,t) -> 32 lanes x 16 halves.
    // lane L, half i  ->  k = i + 16*(L>=16), n = L%16  (joint j = n + 16t).
    for (int e = tid; e < 8192; e += 256) {
        const int i = e & 15;
        const int L = (e >> 4) & 31;
        const int t = (e >> 9) & 1;
        const int s = e >> 10;
        const int j = (L & 15) + 16 * t;
        const int k = i + 16 * (L >> 4);
        const int c = s * 32 + k;
        float v = 0.0f;
        if (j < JJ)
            for (int o = 0; o < CC; ++o) v += w_hm[j * CC + o] * w_proj[o * C2 + c];
        blob[e] = (__bf16)v;
    }
}

// =====================================================================
// Kernel 2: main — fused GEMM (bf16 WMMA) + on-the-fly gaussian GT +
// masked-MSE accumulation + channel-mean accumulation (prefetch pass).
// One block = 128 pixels of one batch image; 8 waves x (16 pix x 32 joints).
// =====================================================================
__global__ void __launch_bounds__(256)
posek_main(const float* __restrict__ rgb, const float* __restrict__ depth,
           unsigned char* __restrict__ ws)
{
    __shared__ __align__(16) unsigned char ldsBlob[16384];  // swizzled w_combo
    __shared__ float ldsBc[JP];
    __shared__ float ldsMask[JP];
    __shared__ int   ldsXc[JP], ldsYc[JP];
    __shared__ float ldsRed[8];

    const int tid   = threadIdx.x;
    const int blk   = blockIdx.x;
    const int b     = blk / 40;
    const int p0blk = (blk % 40) * 128;

    const __bf16* blobG   = (const __bf16*)(ws + WS_BLOB);
    float*        meanc   = (float*)(ws + WS_MEANC);
    float*        lossAcc = (float*)(ws + WS_LOSS);
    const float*  bcombo  = (const float*)(ws + WS_BCOMBO);
    const int*    xcG     = (const int*)(ws + WS_XC);
    const int*    ycG     = (const int*)(ws + WS_YC);
    const float*  mskG    = (const float*)(ws + WS_MASK);

#if USE_TDM
    // --- Tensor Data Mover: DMA the 16KB weight blob into LDS as two 8KB
    // descriptors (wave 0 issues both; waits TENSORcnt before the barrier).
    if (tid < 32) {
        #pragma unroll
        for (int half = 0; half < 2; ++half) {
            const uint32_t lds_off = (uint32_t)(uintptr_t)(&ldsBlob[0]) + 8192u * half;
            const uint64_t ga      = (uint64_t)(uintptr_t)blobG + 8192u * half;
            // D# group0: count=1 | lds_addr | global_addr(57b) | type=2
            v4u g0 = { 1u, lds_off, (uint32_t)ga,
                       (uint32_t)((ga >> 32) & 0x01FFFFFFull) | (2u << 30) };
            // D# group1: data_size=8B, 1-D: dim0=1024, tile_dim0=1024,
            // tensor_dim1=1, dim0_stride=1024 (8-byte units)
            v8i g1 = { (int)(3u << 16),
                       (int)(1024u << 16),   // tensor_dim0[15:0] @ bits 63:48
                       (int)(1u << 16),      // tensor_dim0 hi=0 | tensor_dim1 lo=1
                       (int)(1024u << 16),   // tensor_dim1 hi=0 | tile_dim0=1024
                       0,                    // tile_dim1=0, tile_dim2=0
                       1024,                 // tensor_dim0_stride lo32
                       0, 0 };
            v4i g2 = {0, 0, 0, 0};
            v4i g3 = {0, 0, 0, 0};
#if __clang_major__ >= 23
            v8i g4 = {0, 0, 0, 0, 0, 0, 0, 0};
            __builtin_amdgcn_tensor_load_to_lds(g0, g1, g2, g3, g4, 0);
#else
            __builtin_amdgcn_tensor_load_to_lds(g0, g1, g2, g3, 0);
#endif
        }
    }
#else
    // fallback: cooperative vector copy
    {
        const uint4* src = (const uint4*)blobG;
        uint4*       dst = (uint4*)ldsBlob;
        for (int i = tid; i < 1024; i += 256) dst[i] = src[i];
    }
#endif

    // --- Phase 1: channel-mean accumulation over this 128-pixel slab.
    // Coalesced float4 sweep; also warms L2/WGP$ for the scattered A loads.
    {
        const int c = tid;
        const float* src = (c < CC)
            ? (rgb   + ((size_t)b * CC + c)        * HW + p0blk)
            : (depth + ((size_t)b * CC + (c - CC)) * HW + p0blk);
        const float4* v = (const float4*)src;
        float s = 0.0f;
        #pragma unroll 8
        for (int q = 0; q < 32; ++q) { float4 f = v[q]; s += f.x + f.y + f.z + f.w; }
        atomicAdd(&meanc[b * C2 + c], s);
    }
    if (tid < JP) {
        ldsBc[tid]   = bcombo[tid];
        ldsMask[tid] = mskG[b * JP + tid];
        ldsXc[tid]   = xcG [b * JP + tid];
        ldsYc[tid]   = ycG [b * JP + tid];
    }
#if USE_TDM
    if (tid < 32) __builtin_amdgcn_s_wait_tensorcnt(0);
#endif
    __syncthreads();

    // --- Phase 2: WMMA GEMM. wave -> 16 pixels x 32 joints, K=256 in 8 steps.
    const int L     = tid & 31;
    const int wv    = tid >> 5;
    const int hi    = (L >> 4) & 1;
    const int mLane = L & 15;
    const int p0    = p0blk + wv * 16;

    v8f acc0, acc1;
    {
        const float cb0 = ldsBc[mLane];
        const float cb1 = ldsBc[mLane + 16];
        #pragma unroll
        for (int r = 0; r < 8; ++r) { acc0[r] = cb0; acc1[r] = cb1; }
    }

    const v16bf* bt = (const v16bf*)ldsBlob;
    #pragma unroll
    for (int s = 0; s < 8; ++s) {
        const int cbase = s * 32;
        const float* plane = (cbase < CC)
            ? (rgb   + ((size_t)b * CC + cbase)        * HW)
            : (depth + ((size_t)b * CC + (cbase - CC)) * HW);
        const float* pix = plane + p0 + mLane;
        // A-operand (16x32 bf16), contiguous k = i + 16*hi (matches blob B)
        v16bf a;
        #pragma unroll
        for (int i = 0; i < 16; ++i) {
            const int kloc = i + 16 * hi;
            a[i] = (__bf16)pix[(size_t)kloc * HW];
        }
        const v16bf b0 = bt[(s * 2 + 0) * 32 + L];
        const v16bf b1 = bt[(s * 2 + 1) * 32 + L];
        acc0 = __builtin_amdgcn_wmma_f32_16x16x32_bf16(false, a, false, b0,
                                                       (short)0, acc0, false, false);
        acc1 = __builtin_amdgcn_wmma_f32_16x16x32_bf16(false, a, false, b1,
                                                       (short)0, acc1, false, false);
    }

    // --- Phase 3: masked MSE vs analytic gaussian GT.
    // D element r in lane L  ->  pixel p0 + r + 8*hi, joint (L&15) + 16t.
    const int   j0 = mLane, j1 = mLane + 16;
    const float m0 = ldsMask[j0], m1 = ldsMask[j1];
    const int   x0 = ldsXc[j0], y0 = ldsYc[j0];
    const int   x1 = ldsXc[j1], y1 = ldsYc[j1];
    float local = 0.0f;
    #pragma unroll
    for (int r = 0; r < 8; ++r) {
        const int p  = p0 + 8 * hi + r;
        const int py = p / WW;
        const int px = p - py * WW;
        {
            const int dx = px - x0, dy = py - y0;
            float g = 0.0f;
            if (dx * dx <= 9 && dy * dy <= 9)
                g = expf(-0.5f * (float)(dx * dx + dy * dy));   // sigma = 1
            const float d = acc0[r] - g;
            local += m0 * d * d;
        }
        {
            const int dx = px - x1, dy = py - y1;
            float g = 0.0f;
            if (dx * dx <= 9 && dy * dy <= 9)
                g = expf(-0.5f * (float)(dx * dx + dy * dy));
            const float d = acc1[r] - g;
            local += m1 * d * d;
        }
    }
    // wave-level shuffle reduction, then 8 partials -> one atomic
    #pragma unroll
    for (int off = 16; off > 0; off >>= 1) local += __shfl_down(local, off, 32);
    if (L == 0) ldsRed[wv] = local;
    __syncthreads();
    if (tid == 0) {
        float s = 0.0f;
        #pragma unroll
        for (int i = 0; i < 8; ++i) s += ldsRed[i];
        atomicAdd(lossAcc, s);
    }
}

// =====================================================================
// Kernel 3: finalize — pooled (from channel means, by linearity), presence
// logits, clipped BCE, combine with masked MSE -> scalar loss.
// =====================================================================
__global__ void __launch_bounds__(256)
posek_final(const float* __restrict__ w_proj, const float* __restrict__ b_proj,
            const float* __restrict__ w_pres, const float* __restrict__ b_pres,
            const unsigned char* __restrict__ ws, float* __restrict__ out)
{
    __shared__ float pooled[NB * CC];   // 32 KB
    __shared__ float red0[256], red1[256];
    const float* meanc   = (const float*)(ws + WS_MEANC);
    const float* mskG    = (const float*)(ws + WS_MASK);
    const float* lossAcc = (const float*)(ws + WS_LOSS);
    const int tid = threadIdx.x;
    const float inv = 1.0f / (float)HW;

    for (int idx = tid; idx < NB * CC; idx += 256) {
        const int b = idx >> 7, o = idx & 127;
        float acc = b_proj[o];
        const float* mc = meanc + b * C2;
        for (int c = 0; c < C2; ++c) acc += mc[c] * inv * w_proj[o * C2 + c];
        pooled[idx] = acc;
    }
    __syncthreads();

    float bceL = 0.0f, msL = 0.0f;
    for (int idx = tid; idx < NB * JJ; idx += 256) {
        const int b = idx / JJ, j = idx - b * JJ;
        float lg = b_pres[j];
        const float* po = pooled + b * CC;
        for (int o = 0; o < CC; ++o) lg += po[o] * w_pres[j * CC + o];
        const float p  = 1.0f / (1.0f + expf(-lg));
        const float t  = mskG[b * JP + j];
        const float lp = fmaxf(logf(p), -100.0f);
        const float l1 = fmaxf(logf(1.0f - p), -100.0f);
        bceL += -(t * lp + (1.0f - t) * l1);
        msL  += t;
    }
    red0[tid] = bceL; red1[tid] = msL;
    __syncthreads();
    if (tid == 0) {
        float bs = 0.0f, ms = 0.0f;
        for (int i = 0; i < 256; ++i) { bs += red0[i]; ms += red1[i]; }
        out[0] = lossAcc[0] / (ms + 1e-6f) + bs / (float)(NB * JJ);
    }
}

// =====================================================================
extern "C" void kernel_launch(void* const* d_in, const int* in_sizes, int n_in,
                              void* d_out, int out_size, void* d_ws, size_t ws_size,
                              hipStream_t stream)
{
    (void)in_sizes; (void)n_in; (void)out_size; (void)ws_size;
    const float* rgb    = (const float*)d_in[0];
    const float* depth  = (const float*)d_in[1];
    const float* gt     = (const float*)d_in[2];
    const float* w_proj = (const float*)d_in[3];
    const float* b_proj = (const float*)d_in[4];
    const float* w_hm   = (const float*)d_in[5];
    const float* b_hm   = (const float*)d_in[6];
    const float* w_pres = (const float*)d_in[7];
    const float* b_pres = (const float*)d_in[8];
    unsigned char* ws   = (unsigned char*)d_ws;

    posek_prep <<<1,       256, 0, stream>>>(gt, w_proj, b_proj, w_hm, b_hm, ws);
    posek_main <<<NB * 40, 256, 0, stream>>>(rgb, depth, ws);
    posek_final<<<1,       256, 0, stream>>>(w_proj, b_proj, w_pres, b_pres, ws,
                                             (float*)d_out);
}
